// RNN_2525440770480
// MI455X (gfx1250) — matrix operand order repaired and checked by reading, not compiled, not run on hardware
//
#include <hip/hip_runtime.h>
#include <cmath>

// ---------------- problem constants ----------------
#define T_STEPS 256
#define BATCH   64
#define HID     1024           // I == H == 1024
#define BH      (BATCH * HID)  // 65536
#define NBLK    32
#define NTHREADS 256
#define ACT_BYTES (BATCH * HID * 2)   // one 64x1024 bf16 activation matrix: 128 KB
#define SH_BYTES  (2 * ACT_BYTES)     // both activation operands in LDS: 256 KB (< 320 KB/WGP)

typedef __attribute__((ext_vector_type(8)))  __bf16 v8bf;
typedef __attribute__((ext_vector_type(16))) __bf16 v16bf;
typedef __attribute__((ext_vector_type(8)))  float  v8f;

// --------- WMMA fragment loaders (CDNA5 ISA 7.12.2 layouts, wave32) ---------
// A 16x32 bf16: lane m (0-15): VGPR0-3 = K {k..k+7}, VGPR4-7 = K {k+16..k+23}
//               lane m+16:     VGPR0-3 = K {k+8..k+15}, VGPR4-7 = K {k+24..k+31}
__device__ __forceinline__ v16bf ldfragA(const __bf16* rowp, int k, int hi) {
  union { v16bf v; v8bf h[2]; } u;
  const __bf16* p = rowp + k + hi * 8;
  u.h[0] = *(const v8bf*)p;
  u.h[1] = *(const v8bf*)(p + 16);
  return u.v;
}
// B 32x16 bf16: lane n (0-15) holds column n, K {k..k+7, k+16..k+23}; lane n+16: K {k+8..k+15, k+24..k+31}
__device__ __forceinline__ v16bf ldfragB(const __bf16* rowp, int k, int hi) {
  union { v16bf v; v8bf h[2]; } u;
  const __bf16* p = rowp + k + hi * 8;
  u.h[0] = *(const v8bf*)p;
  u.h[1] = *(const v8bf*)(p + 16);
  return u.v;
}

// ---------------- CDNA5 async global->LDS staging (ASYNCcnt) ----------------
__device__ __forceinline__ void async_b128(unsigned lds_off, const void* gptr) {
  asm volatile("global_load_async_to_lds_b128 %0, %1, off"
               :: "v"(lds_off), "v"(gptr) : "memory");
}
__device__ __forceinline__ void wait_async0() {
  asm volatile("s_wait_asynccnt 0x0" ::: "memory");
}

// Stage both activation matrices (64x1024 bf16 each) into LDS cooperatively.
// 8192 x 16B chunks per matrix / 256 threads = 32 async b128 per thread per matrix.
__device__ __forceinline__ void stage_acts(const __bf16* act1, const __bf16* act2,
                                           unsigned sh_base, int tid) {
#pragma unroll 4
  for (int i = 0; i < 32; ++i) {
    unsigned c = (unsigned)(i * NTHREADS + tid) * 16u;   // byte offset of 16B chunk
    async_b128(sh_base + c, (const char*)act1 + c);
  }
#pragma unroll 4
  for (int i = 0; i < 32; ++i) {
    unsigned c = (unsigned)(i * NTHREADS + tid) * 16u;
    async_b128(sh_base + ACT_BYTES + c, (const char*)act2 + c);
  }
  wait_async0();       // our wave's async copies landed in LDS
  __syncthreads();     // every wave's copies landed
}

// One layer-phase GEMM for this wave's 16x16 tile:
// A fragments from LDS (ds_load_b128), B (weights) streamed from L2.
__device__ __forceinline__ v8f gemm_pair(const __bf16* aRow1, const __bf16* bRow1,
                                         const __bf16* aRow2, const __bf16* bRow2,
                                         int hi) {
  v8f acc = {};
#pragma unroll 4
  for (int k = 0; k < HID; k += 32) {
    v16bf a = ldfragA(aRow1, k, hi);
    v16bf b = ldfragB(bRow1, k, hi);
    acc = __builtin_amdgcn_wmma_f32_16x16x32_bf16(false, a, false, b, (short)0, acc,
                                                  false, false);
  }
#pragma unroll 4
  for (int k = 0; k < HID; k += 32) {
    v16bf a = ldfragA(aRow2, k, hi);
    v16bf b = ldfragB(bRow2, k, hi);
    acc = __builtin_amdgcn_wmma_f32_16x16x32_bf16(false, a, false, b, (short)0, acc,
                                                  false, false);
  }
  return acc;
}

// ---------------- device-wide split barrier (persistent kernel) ----------------
__device__ __forceinline__ void grid_barrier(int* bar, int target) {
  __builtin_amdgcn_fence(__ATOMIC_RELEASE, "agent");   // global_wb: publish h-state
  __syncthreads();
  if (threadIdx.x == 0) {
    __hip_atomic_fetch_add(bar, 1, __ATOMIC_RELAXED, __HIP_MEMORY_SCOPE_AGENT);
    while (__hip_atomic_load(bar, __ATOMIC_RELAXED, __HIP_MEMORY_SCOPE_AGENT) < target) {
      __builtin_amdgcn_s_sleep(2);
    }
  }
  __syncthreads();
  __builtin_amdgcn_fence(__ATOMIC_ACQUIRE, "agent");   // global_inv: see peers' h-state
}

// ---------------- persistent 2-layer RNN kernel ----------------
__global__ __launch_bounds__(NTHREADS, 1) void rnn_persistent(
    const __bf16* __restrict__ xb,
    const __bf16* __restrict__ wih0, const __bf16* __restrict__ whh0,
    const __bf16* __restrict__ wih1, const __bf16* __restrict__ whh1,
    const float* __restrict__ bsum0, const float* __restrict__ bsum1,
    __bf16* h0s, __bf16* h1s,
    float* __restrict__ out, int* bar) {
  extern __shared__ __bf16 shact[];                 // [2][64][1024] bf16
  const __bf16* shA1 = shact;                       // staged act operand 1
  const __bf16* shA2 = shact + (size_t)BATCH * HID; // staged act operand 2
  const unsigned sh_base = (unsigned)(uintptr_t)&shact[0];  // LDS byte offset (addr[31:0])

  const int tid   = threadIdx.x;
  const int lane  = tid & 31;
  const int wave  = tid >> 5;
  const int mt    = wave & 3;       // M tile 0..3 (batch rows 16*mt..)
  const int nc    = wave >> 2;      // 0..1
  const int r     = lane & 15;
  const int hi    = lane >> 4;
  const long arow = mt * 16 + r;                  // activation (batch) row for A frag
  const int  ncol = blockIdx.x * 32 + nc * 16;    // output column base
  const long brow = ncol + r;                     // weight row for B frag
  const int  n    = ncol + r;                     // output column for epilogue
  const int  mbase = mt * 16 + hi * 8;            // output row base for epilogue
  const float bs0 = bsum0[n];
  const float bs1 = bsum1[n];
  float* fin0 = out + (size_t)T_STEPS * BH;       // h_n[0]
  float* fin1 = fin0 + BH;                        // h_n[1]

  int phase = 0;
  for (int t = 0; t < T_STEPS; ++t) {
    const __bf16* h0o = h0s + (size_t)(t & 1) * BH;
    __bf16*       h0n = h0s + (size_t)((t + 1) & 1) * BH;
    const __bf16* h1o = h1s + (size_t)(t & 1) * BH;
    __bf16*       h1n = h1s + (size_t)((t + 1) & 1) * BH;
    const __bf16* xt  = xb + (size_t)t * BH;      // B*I == BH

    // ---- layer 0: h0n = tanh(x_t @ Wih0^T + h0o @ Whh0^T + bias) ----
    stage_acts(xt, h0o, sh_base, tid);
    v8f acc = gemm_pair(shA1 + arow * HID, wih0 + brow * HID,
                        shA2 + arow * HID, whh0 + brow * HID, hi);
#pragma unroll
    for (int v = 0; v < 8; ++v) {
      float val = tanhf(acc[v] + bs0);
      int row = mbase + v;
      h0n[(size_t)row * HID + n] = (__bf16)val;
      if (t == T_STEPS - 1) fin0[(size_t)row * HID + n] = val;
    }
    grid_barrier(bar, ++phase * NBLK);

    // ---- layer 1: h1n = tanh(h0n @ Wih1^T + h1o @ Whh1^T + bias) ----
    stage_acts(h0n, h1o, sh_base, tid);
    acc = gemm_pair(shA1 + arow * HID, wih1 + brow * HID,
                    shA2 + arow * HID, whh1 + brow * HID, hi);
    float* outt = out + (size_t)t * BH;
#pragma unroll
    for (int v = 0; v < 8; ++v) {
      float val = tanhf(acc[v] + bs1);
      int row = mbase + v;
      h1n[(size_t)row * HID + n] = (__bf16)val;
      outt[(size_t)row * HID + n] = val;
      if (t == T_STEPS - 1) fin1[(size_t)row * HID + n] = val;
    }
    grid_barrier(bar, ++phase * NBLK);
  }
}

// ---------------- prep kernels ----------------
__global__ void f32_to_bf16(const float* __restrict__ s, __bf16* __restrict__ d, int n) {
  int i = blockIdx.x * blockDim.x + threadIdx.x;
  if (i < n) d[i] = (__bf16)s[i];
}
__global__ void bias_sum(const float* __restrict__ a, const float* __restrict__ b,
                         float* __restrict__ d, int n) {
  int i = blockIdx.x * blockDim.x + threadIdx.x;
  if (i < n) d[i] = a[i] + b[i];
}

// ---------------- host launch ----------------
extern "C" void kernel_launch(void* const* d_in, const int* in_sizes, int n_in,
                              void* d_out, int out_size, void* d_ws, size_t ws_size,
                              hipStream_t stream) {
  const float* x    = (const float*)d_in[0];
  const float* h0in = (const float*)d_in[1];
  const float* Wih0 = (const float*)d_in[2];
  const float* bih0 = (const float*)d_in[3];
  const float* Whh0 = (const float*)d_in[4];
  const float* bhh0 = (const float*)d_in[5];
  const float* Wih1 = (const float*)d_in[6];
  const float* bih1 = (const float*)d_in[7];
  const float* Whh1 = (const float*)d_in[8];
  const float* bhh1 = (const float*)d_in[9];
  float* out = (float*)d_out;

  uint8_t* ws   = (uint8_t*)d_ws;
  int*   bar    = (int*)ws;                      // [0, 256)
  float* bsum0  = (float*)(ws + 256);            // H floats
  float* bsum1  = bsum0 + HID;
  __bf16* xb    = (__bf16*)(ws + 16384);         // T*B*I bf16 (32 MB)
  __bf16* wih0b = xb    + (size_t)T_STEPS * BH;
  __bf16* whh0b = wih0b + (size_t)HID * HID;
  __bf16* wih1b = whh0b + (size_t)HID * HID;
  __bf16* whh1b = wih1b + (size_t)HID * HID;
  __bf16* h0s   = whh1b + (size_t)HID * HID;     // 2 ping-pong buffers
  __bf16* h1s   = h0s + 2 * (size_t)BH;          // 2 ping-pong buffers
  (void)ws_size; (void)in_sizes; (void)n_in; (void)out_size;

  hipMemsetAsync(bar, 0, 256, stream);           // reset barrier counter every call

  const int nx = T_STEPS * BH;
  const int nw = HID * HID;
  f32_to_bf16<<<(nx + 255) / 256, 256, 0, stream>>>(x, xb, nx);
  f32_to_bf16<<<(nw + 255) / 256, 256, 0, stream>>>(Wih0, wih0b, nw);
  f32_to_bf16<<<(nw + 255) / 256, 256, 0, stream>>>(Whh0, whh0b, nw);
  f32_to_bf16<<<(nw + 255) / 256, 256, 0, stream>>>(Wih1, wih1b, nw);
  f32_to_bf16<<<(nw + 255) / 256, 256, 0, stream>>>(Whh1, whh1b, nw);
  f32_to_bf16<<<(BH + 255) / 256, 256, 0, stream>>>(h0in, h0s, BH);
  f32_to_bf16<<<(BH + 255) / 256, 256, 0, stream>>>(h0in + BH, h1s, BH);
  bias_sum<<<(HID + 255) / 256, 256, 0, stream>>>(bih0, bhh0, bsum0, HID);
  bias_sum<<<(HID + 255) / 256, 256, 0, stream>>>(bih1, bhh1, bsum1, HID);

  rnn_persistent<<<NBLK, NTHREADS, SH_BYTES, stream>>>(xb, wih0b, whh0b, wih1b, whh1b,
                                                       bsum0, bsum1, h0s, h1s, out, bar);
}
